// GAT_86964497809912
// MI455X (gfx1250) — matrix operand order repaired and checked by reading, not compiled, not run on hardware
//
#include <hip/hip_runtime.h>
#include <stdint.h>

// ---------------------------------------------------------------------------
// GAT for MI455X (gfx1250, wave32, WMMA).  All GEMMs via v_wmma_f32_16x16x32_f16.
// Flash-style online-softmax attention: never materializes the NxN logits.
// ---------------------------------------------------------------------------

typedef __attribute__((ext_vector_type(16))) _Float16 v16h;
typedef __attribute__((ext_vector_type(8)))  _Float16 v8h;
typedef __attribute__((ext_vector_type(8)))  float    v8f;

#define NNODE  4096
#define NFEAT  512
#define NHID   64
#define NHEADS 8
#define NPAIR  65536
#define ALPHA  0.2f
#define NEGV   -9.0e15f

// ---------------- small conversion / packing kernels ------------------------

__global__ void k_f32_to_f16(const float* __restrict__ src,
                             _Float16* __restrict__ dst, int n) {
  int i = blockIdx.x * 256 + threadIdx.x;
  if (i < n) dst[i] = (_Float16)src[i];
}

// W_heads [8,512,64] -> WT1 [512(n=h*64+j)][512(k)] f16 (B-transposed for WMMA)
__global__ void k_pack_W1(const float* __restrict__ Wheads,
                          _Float16* __restrict__ WT1) {
  int o = blockIdx.x * 256 + threadIdx.x;   // o < 512*512
  int n = o >> 9, k = o & 511;
  int h = n >> 6, j = n & 63;
  WT1[o] = (_Float16)Wheads[((size_t)(h * NFEAT + k)) * NHID + j];
}

// W_out [512,64] -> WoutT [64(n)][512(k)] f16
__global__ void k_pack_Wout(const float* __restrict__ Wout,
                            _Float16* __restrict__ WoutT) {
  int o = blockIdx.x * 256 + threadIdx.x;   // o < 64*512
  int n = o >> 9, k = o & 511;
  WoutT[o] = (_Float16)Wout[(size_t)k * NHID + n];
}

// ---------------- WMMA GEMM: out = A[Mtot x K] * B (BT is [Ntot x K]) -------
// One wave computes a 16x64 tile.  Writes f32 row-major and f16 transposed.

__global__ __launch_bounds__(32)
void k_gemm_f16(const _Float16* __restrict__ A,
                const _Float16* __restrict__ BT,
                float* __restrict__ outF,      // [Mtot x Ntot] f32
                _Float16* __restrict__ outT,   // [Ntot x Mtot] f16 (transposed)
                int K, int Mtot) {
  const int l    = threadIdx.x;
  const int half = l >> 4;
  const int r    = l & 15;
  const int Ntot = gridDim.y * 64;
  const int row  = blockIdx.x * 16 + r;
  const int colb = blockIdx.y * 64;

  const _Float16* arow = A + (size_t)row * K;
  v8f acc0 = {}, acc1 = {}, acc2 = {}, acc3 = {};

  for (int k0 = 0; k0 < K; k0 += 32) {
    union { v16h v; v8h h[2]; } af;
    af.h[0] = *(const v8h*)(arow + k0 + half * 16);       // K = half*16 + 0..7
    af.h[1] = *(const v8h*)(arow + k0 + half * 16 + 8);   // K = half*16+8+0..7
    const int kb = k0 + half * 16;
    v16h b0 = *(const v16h*)(BT + (size_t)(colb + 0  + r) * K + kb);
    v16h b1 = *(const v16h*)(BT + (size_t)(colb + 16 + r) * K + kb);
    v16h b2 = *(const v16h*)(BT + (size_t)(colb + 32 + r) * K + kb);
    v16h b3 = *(const v16h*)(BT + (size_t)(colb + 48 + r) * K + kb);
    acc0 = __builtin_amdgcn_wmma_f32_16x16x32_f16(false, af.v, false, b0, (short)0, acc0, false, false);
    acc1 = __builtin_amdgcn_wmma_f32_16x16x32_f16(false, af.v, false, b1, (short)0, acc1, false, false);
    acc2 = __builtin_amdgcn_wmma_f32_16x16x32_f16(false, af.v, false, b2, (short)0, acc2, false, false);
    acc3 = __builtin_amdgcn_wmma_f32_16x16x32_f16(false, af.v, false, b3, (short)0, acc3, false, false);
  }

  #pragma unroll
  for (int t = 0; t < 4; ++t) {
    v8f acc = (t == 0) ? acc0 : (t == 1) ? acc1 : (t == 2) ? acc2 : acc3;
    const int col = colb + t * 16 + r;
    #pragma unroll
    for (int rr = 0; rr < 8; ++rr) {
      const int M = rr + 8 * half;
      const float v = acc[rr];
      outF[(size_t)(blockIdx.x * 16 + M) * Ntot + col] = v;
      outT[(size_t)col * Mtot + (blockIdx.x * 16 + M)] = (_Float16)v;
    }
  }
}

// ---------------- e_src / e_dst : Wh . a ------------------------------------

__global__ void k_edge_logits(const float* __restrict__ Wh,   // [N x lda] f32
                              const float* __restrict__ a,    // [nheads x 2*NHID]
                              float* __restrict__ esrc, float* __restrict__ edst,
                              int lda, int nheads) {
  int i = blockIdx.x * 256 + threadIdx.x;
  if (i >= nheads * NNODE) return;
  int h = i >> 12, n = i & (NNODE - 1);
  const float* w  = Wh + (size_t)n * lda + h * NHID;
  const float* av = a + (size_t)h * 2 * NHID;
  float s0 = 0.f, s1 = 0.f;
  #pragma unroll
  for (int j = 0; j < NHID; ++j) {
    s0 = fmaf(w[j], av[j], s0);
    s1 = fmaf(w[j], av[NHID + j], s1);
  }
  esrc[i] = s0;
  edst[i] = s1;
}

// ---------------- flash-style GAT attention ---------------------------------
// One wave per 16-row block per head.  Online softmax over 4096 columns in
// K=32 tiles; probabilities are built directly in the WMMA A-fragment layout.

__global__ __launch_bounds__(32)
void k_gat_attn(const float* __restrict__ adj,
                const float* __restrict__ esrc_all,
                const float* __restrict__ edst_all,
                const _Float16* __restrict__ WhT_all,  // [(h*64+n) x 4096] f16
                float* __restrict__ outF,              // nullable, ld = heads*64
                _Float16* __restrict__ outH) {         // nullable, ld = heads*64
  const int l    = threadIdx.x;
  const int half = l >> 4;
  const int r    = l & 15;
  const int h    = blockIdx.y;
  const int ld   = gridDim.y * 64;
  const int row  = blockIdx.x * 16 + r;

  const float*    es   = esrc_all + (size_t)h * NNODE;
  const float*    ed   = edst_all + (size_t)h * NNODE;
  const _Float16* WhT  = WhT_all + (size_t)h * 64 * NNODE;
  const float*    arow = adj + (size_t)row * NNODE;
  const float     esv  = es[row];

  __shared__ float s_max[16], s_sum[16], s_scale[16];
  if (l < 16) { s_max[l] = -3.0e38f; s_sum[l] = 0.f; }
  __syncthreads();

  v8f acc0 = {}, acc1 = {}, acc2 = {}, acc3 = {};

  auto lg = [&](float av, float dv) -> float {
    float t  = esv + dv;
    float lv = t > 0.f ? t : ALPHA * t;
    return av > 0.f ? lv : NEGV;
  };

  for (int j0 = 0; j0 < NNODE; j0 += 32) {
    const int c0 = j0 + half * 16;       // this lane's K = half*16 + 0..7
    const int c1 = j0 + half * 16 + 8;   // this lane's K = half*16+8 + 0..7
    float4 a0 = *(const float4*)(arow + c0);
    float4 a1 = *(const float4*)(arow + c0 + 4);
    float4 a2 = *(const float4*)(arow + c1);
    float4 a3 = *(const float4*)(arow + c1 + 4);
    float4 d0 = *(const float4*)(ed + c0);
    float4 d1 = *(const float4*)(ed + c0 + 4);
    float4 d2 = *(const float4*)(ed + c1);
    float4 d3 = *(const float4*)(ed + c1 + 4);

    float m[16];
    m[0]=lg(a0.x,d0.x);  m[1]=lg(a0.y,d0.y);  m[2]=lg(a0.z,d0.z);  m[3]=lg(a0.w,d0.w);
    m[4]=lg(a1.x,d1.x);  m[5]=lg(a1.y,d1.y);  m[6]=lg(a1.z,d1.z);  m[7]=lg(a1.w,d1.w);
    m[8]=lg(a2.x,d2.x);  m[9]=lg(a2.y,d2.y);  m[10]=lg(a2.z,d2.z); m[11]=lg(a2.w,d2.w);
    m[12]=lg(a3.x,d3.x); m[13]=lg(a3.y,d3.y); m[14]=lg(a3.z,d3.z); m[15]=lg(a3.w,d3.w);

    float lmax = m[0];
    #pragma unroll
    for (int e = 1; e < 16; ++e) lmax = fmaxf(lmax, m[e]);
    float tmax = fmaxf(lmax, __shfl_xor(lmax, 16, 32));  // combine half-waves

    float oldm = s_max[r];
    float nm   = fmaxf(oldm, tmax);
    float sc   = __expf(oldm - nm);

    float p[16];
    float lsum = 0.f;
    #pragma unroll
    for (int e = 0; e < 16; ++e) { p[e] = __expf(m[e] - nm); lsum += p[e]; }
    float tsum = lsum + __shfl_xor(lsum, 16, 32);

    if (l < 16) { s_max[l] = nm; s_sum[l] = s_sum[l] * sc + tsum; s_scale[l] = sc; }
    __syncthreads();

    // rescale accumulators by per-row exp(old_max - new_max)
    #pragma unroll
    for (int rr = 0; rr < 8; ++rr) {
      const float scm = s_scale[rr + 8 * half];
      acc0[rr] *= scm; acc1[rr] *= scm; acc2[rr] *= scm; acc3[rr] *= scm;
    }

    // probabilities are already in the 16x32 f16 A-fragment layout
    v16h af;
    #pragma unroll
    for (int e = 0; e < 16; ++e) af[e] = (_Float16)p[e];

    const int kb = j0 + half * 16;
    v16h b0 = *(const v16h*)(WhT + (size_t)(0  + r) * NNODE + kb);
    v16h b1 = *(const v16h*)(WhT + (size_t)(16 + r) * NNODE + kb);
    v16h b2 = *(const v16h*)(WhT + (size_t)(32 + r) * NNODE + kb);
    v16h b3 = *(const v16h*)(WhT + (size_t)(48 + r) * NNODE + kb);
    acc0 = __builtin_amdgcn_wmma_f32_16x16x32_f16(false, af, false, b0, (short)0, acc0, false, false);
    acc1 = __builtin_amdgcn_wmma_f32_16x16x32_f16(false, af, false, b1, (short)0, acc1, false, false);
    acc2 = __builtin_amdgcn_wmma_f32_16x16x32_f16(false, af, false, b2, (short)0, acc2, false, false);
    acc3 = __builtin_amdgcn_wmma_f32_16x16x32_f16(false, af, false, b3, (short)0, acc3, false, false);
    __syncthreads();
  }

  // epilogue: divide by softmax denom, ELU, store
  #pragma unroll
  for (int t = 0; t < 4; ++t) {
    v8f acc = (t == 0) ? acc0 : (t == 1) ? acc1 : (t == 2) ? acc2 : acc3;
    const int col = h * 64 + t * 16 + r;
    #pragma unroll
    for (int rr = 0; rr < 8; ++rr) {
      const int M = rr + 8 * half;
      float v = acc[rr] / s_sum[M];
      v = v > 0.f ? v : (__expf(v) - 1.0f);   // ELU
      const int orow = blockIdx.x * 16 + M;
      if (outF) outF[(size_t)orow * ld + col] = v;
      if (outH) outH[(size_t)orow * ld + col] = (_Float16)v;
    }
  }
}

// ---------------- pair scores: diag(E1 @ W_score @ E2^T) --------------------

__global__ __launch_bounds__(256)
void k_score(const float* __restrict__ h2, const float* __restrict__ Wscore,
             const int* __restrict__ p1, const int* __restrict__ p2,
             float* __restrict__ out, int P) {
  __shared__ float Ws[64 * 64];
  for (int i = threadIdx.x; i < 64 * 64; i += 256) Ws[i] = Wscore[i];
  __syncthreads();
  int p = blockIdx.x * 256 + threadIdx.x;
  if (p >= P) return;
  const float* t = h2 + (size_t)p1[p] * NHID;
  const float* u = h2 + (size_t)p2[p] * NHID;
  float ur[64];
  #pragma unroll
  for (int j = 0; j < 64; ++j) ur[j] = u[j];
  float s = 0.f;
  for (int i = 0; i < 64; ++i) {
    float acc = 0.f;
    #pragma unroll
    for (int j = 0; j < 64; ++j) acc = fmaf(Ws[i * 64 + j], ur[j], acc);
    s = fmaf(t[i], acc, s);
  }
  out[p] = s;
}

// ---------------------------------------------------------------------------

extern "C" void kernel_launch(void* const* d_in, const int* in_sizes, int n_in,
                              void* d_out, int out_size, void* d_ws, size_t ws_size,
                              hipStream_t stream) {
  (void)in_sizes; (void)n_in; (void)out_size; (void)ws_size;
  const float* x      = (const float*)d_in[0];
  const float* adj    = (const float*)d_in[1];
  const float* Wheads = (const float*)d_in[2];
  const float* aheads = (const float*)d_in[3];
  const float* Wout   = (const float*)d_in[4];
  const float* aout   = (const float*)d_in[5];
  const float* Wscore = (const float*)d_in[6];
  const int*   p1     = (const int*)d_in[7];
  const int*   p2     = (const int*)d_in[8];
  float* scores = (float*)d_out;

  char* w = (char*)d_ws;
  size_t off = 0;
  auto alloc = [&](size_t bytes) -> char* {
    char* p = w + off;
    off = (off + bytes + 255) & ~(size_t)255;
    return p;
  };
  _Float16* xh    = (_Float16*)alloc((size_t)NNODE * NFEAT * 2);
  _Float16* WT1   = (_Float16*)alloc((size_t)512 * 512 * 2);
  _Float16* WoutT = (_Float16*)alloc((size_t)64 * 512 * 2);
  float*    Wh1   = (float*)   alloc((size_t)NNODE * 512 * 4);
  _Float16* WhT1  = (_Float16*)alloc((size_t)512 * NNODE * 2);
  float*    e1s   = (float*)   alloc((size_t)NHEADS * NNODE * 4);
  float*    e1d   = (float*)   alloc((size_t)NHEADS * NNODE * 4);
  _Float16* h1h   = (_Float16*)alloc((size_t)NNODE * 512 * 2);
  float*    Wh2   = (float*)   alloc((size_t)NNODE * 64 * 4);
  _Float16* WhT2  = (_Float16*)alloc((size_t)64 * NNODE * 2);
  float*    e2s   = (float*)   alloc((size_t)NNODE * 4);
  float*    e2d   = (float*)   alloc((size_t)NNODE * 4);
  float*    h2    = (float*)   alloc((size_t)NNODE * 64 * 4);

  // input prep (f32 -> f16, B-transposed weight layouts)
  k_f32_to_f16<<<(NNODE * NFEAT + 255) / 256, 256, 0, stream>>>(x, xh, NNODE * NFEAT);
  k_pack_W1  <<<(512 * 512) / 256, 256, 0, stream>>>(Wheads, WT1);
  k_pack_Wout<<<(64 * 512) / 256, 256, 0, stream>>>(Wout, WoutT);

  // layer 1: Wh = x @ Wcat (4096x512x512), then per-head edge logits + attention
  k_gemm_f16<<<dim3(NNODE / 16, 8), 32, 0, stream>>>(xh, WT1, Wh1, WhT1, NFEAT, NNODE);
  k_edge_logits<<<(NHEADS * NNODE) / 256, 256, 0, stream>>>(Wh1, aheads, e1s, e1d, 512, NHEADS);
  k_gat_attn<<<dim3(NNODE / 16, NHEADS), 32, 0, stream>>>(adj, e1s, e1d, WhT1, nullptr, h1h);

  // layer 2: Wh2 = h1 @ W_out (4096x512x64), edge logits, attention (+ELU)
  k_gemm_f16<<<dim3(NNODE / 16, 1), 32, 0, stream>>>(h1h, WoutT, Wh2, WhT2, 512, NNODE);
  k_edge_logits<<<NNODE / 256, 256, 0, stream>>>(Wh2, aout, e2s, e2d, 64, 1);
  k_gat_attn<<<dim3(NNODE / 16, 1), 32, 0, stream>>>(adj, e2s, e2d, WhT2, h2, nullptr);

  // pair scores
  k_score<<<NPAIR / 256, 256, 0, stream>>>(h2, Wscore, p1, p2, scores, NPAIR);
}
